// StaticAtomicModule_48885317763465
// MI455X (gfx1250) — compile-verified, matrix operand order checked by reading, not run on hardware
//
#include <hip/hip_runtime.h>

// ---------------------------------------------------------------------------
// ANI-style species MLP on MI455X (gfx1250), f16 WMMA + f32 accumulate.
// Weights staged into LDS via CDNA5 async-to-LDS loads (ASYNCcnt),
// double-buffered per K-slice; 8 waves = 4 m-groups x 2 n-halves so each
// B fragment read from LDS feeds two WMMAs.
// Output is a single scalar: sum over all atoms of per-atom energies.
// ---------------------------------------------------------------------------

typedef __attribute__((ext_vector_type(16))) _Float16 v16h;
typedef __attribute__((ext_vector_type(8)))  float    v8f;
typedef __attribute__((ext_vector_type(4)))  unsigned int v4u;

#define AEV_DIM 384
#define H1D 160
#define H2D 128
#define H3D 96
#define NSPEC 4
#define CELU_ALPHA 0.1f
#define MTILE 128

// LDS strides in f16 units; dword stride odd => conflict-free ds_load_b32
#define STRIDE_A  390   // 195 dwords (odd)
#define STRIDE_H1 166   //  83 dwords (odd)
#define STRIDE_H2 134   //  67 dwords (odd)
#define STRIDE_H3 102   //  51 dwords (odd)

#define NT1 (H1D/16)      // 10
#define KS1 (AEV_DIM/32)  // 12
#define NT2 (H2D/16)      // 8
#define KS2 (H1D/32)      // 5
#define NT3 (H3D/16)      // 6
#define KS3 (H2D/32)      // 4
#define FRAG_DW 256       // 32 lanes * 8 dwords per 16x32 f16 fragment

#define DW_W1 (NSPEC*NT1*KS1*FRAG_DW)
#define DW_W2 (NSPEC*NT2*KS2*FRAG_DW)
#define DW_W3 (NSPEC*NT3*KS3*FRAG_DW)

#define WBUF_DW (2 * NT1 * FRAG_DW)   // double buffer sized for largest slice (2x10KB)

// ---- CDNA5 async-to-LDS path (guarded; falls back to sync vector copy) ----
#if defined(__has_builtin)
#  if __has_builtin(__builtin_amdgcn_global_load_async_to_lds_b128)
#    define HAVE_ASYNC 1
#  endif
#  if __has_builtin(__builtin_amdgcn_s_wait_asynccnt)
#    define HAVE_WAIT_ASYNC 1
#  endif
#endif
#ifndef HAVE_ASYNC
#  define HAVE_ASYNC 0
#endif
#ifndef HAVE_WAIT_ASYNC
#  define HAVE_WAIT_ASYNC 0
#endif

union FragU {
  v16h h;
  unsigned int u[8];
  v4u  q[2];
};

__device__ __forceinline__ float celu_f(float x) {
  return x > 0.f ? x : CELU_ALPHA * (__expf(x * (1.0f / CELU_ALPHA)) - 1.0f);
}

// Copy one 16B chunk global -> LDS (async if available).
__device__ __forceinline__ void copy16(unsigned int* __restrict__ ldsdst,
                                       const unsigned int* __restrict__ gsrc) {
#if HAVE_ASYNC
  typedef int v4i_gcc __attribute__((vector_size(16)));
  typedef __attribute__((address_space(1))) v4i_gcc as1_v4i;
  typedef __attribute__((address_space(3))) v4i_gcc as3_v4i;
  __builtin_amdgcn_global_load_async_to_lds_b128((as1_v4i*)gsrc, (as3_v4i*)ldsdst, 0, 0);
#else
  *(v4u*)ldsdst = *(const v4u*)gsrc;
#endif
}

// Copy N16 16-byte chunks global -> LDS, 256 threads cooperating, fully
// unrolled (N16 is a compile-time constant: no dynamic loop control).
template <int N16>
__device__ __forceinline__ void slice_copy(unsigned int* __restrict__ ldsdst,
                                           const unsigned int* __restrict__ gsrc,
                                           int tid) {
  constexpr int FULL = N16 / 256;
  constexpr int REM  = N16 % 256;
#pragma unroll
  for (int r = 0; r < FULL; ++r) {
    const int i = tid + r * 256;
    copy16(ldsdst + (size_t)i * 4, gsrc + (size_t)i * 4);
  }
  if (REM > 0 && tid < REM) {
    const int i = tid + FULL * 256;
    copy16(ldsdst + (size_t)i * 4, gsrc + (size_t)i * 4);
  }
}

// Wait for this wave's outstanding async copies, then block barrier.
__device__ __forceinline__ void slice_sync() {
#if HAVE_ASYNC
#  if HAVE_WAIT_ASYNC
  __builtin_amdgcn_s_wait_asynccnt(0);
#  else
  asm volatile("s_wait_asynccnt 0x0" ::: "memory");
#  endif
#endif
  __syncthreads();
}

// ---------------------------------------------------------------------------
// Prep: repack fp32 row-major W[K][N] (per species) into f16 B fragments for
// v_wmma_f32_16x16x32_f16, ks-major so each K-slice is contiguous (for the
// async slice copies). Assumed B layout = mirror of documented 16-bit A layout:
//   lane l holds column n=(l&15); kb=(l>=16)?8:0
//   vgpr j<4 : K = kb+2j,2j+1 ; vgpr j>=4 : K = kb+16+2(j-4),+1
// Packed dword index: (((spec*KS + ks)*NT + nt)*32 + lane)*8 + j
// ---------------------------------------------------------------------------
__global__ void pack_weights_kernel(const float* __restrict__ W1,
                                    const float* __restrict__ W2,
                                    const float* __restrict__ W3,
                                    unsigned int* __restrict__ P1,
                                    unsigned int* __restrict__ P2,
                                    unsigned int* __restrict__ P3) {
  int id = blockIdx.x * 256 + threadIdx.x;
  const float* W; unsigned int* P; int NT, KS, N;
  if (id < DW_W1)                  { W = W1; P = P1; NT = NT1; KS = KS1; N = H1D; }
  else if (id < DW_W1 + DW_W2)     { id -= DW_W1;         W = W2; P = P2; NT = NT2; KS = KS2; N = H2D; }
  else if (id < DW_W1+DW_W2+DW_W3) { id -= (DW_W1+DW_W2); W = W3; P = P3; NT = NT3; KS = KS3; N = H3D; }
  else return;

  const int pid  = id;
  const int j    = id & 7;
  const int lane = (id >> 3) & 31;
  int rest = id >> 8;
  const int nt = rest % NT; rest /= NT;
  const int ks = rest % KS;
  const int s  = rest / KS;
  const int K  = KS * 32;

  const int n   = nt * 16 + (lane & 15);
  const int kb  = (lane >> 4) * 8;
  const int klo = (j < 4) ? (kb + 2 * j) : (kb + 16 + 2 * (j - 4));
  const int k0  = ks * 32 + klo;

  const float* Ws = W + (size_t)s * K * N;
  union { _Float16 f[2]; unsigned int u; } pk;
  pk.f[0] = (_Float16)Ws[(size_t)(k0    ) * N + n];
  pk.f[1] = (_Float16)Ws[(size_t)(k0 + 1) * N + n];
  P[pid] = pk.u;
}

// ---------------------------------------------------------------------------
// One WMMA layer. Wave split: mg = wave&3 owns rows [mg*32, mg*32+32) as two
// 16-row sub-tiles; nh = wave>>2 owns half of the N tiles. Each B fragment is
// loaded once (2x ds_load_b128, immediate offsets) and feeds two WMMAs.
// ---------------------------------------------------------------------------
template <int NT, int KS, int SSTR, int DSTR>
__device__ __forceinline__ void mlp_layer(const _Float16* __restrict__ src,
                                          _Float16* __restrict__ dst,
                                          const unsigned int* __restrict__ pw,
                                          unsigned int* __restrict__ wbuf,
                                          const float* __restrict__ bias,
                                          int wave, int lane, int tid) {
  constexpr int NTH = NT / 2;

  // stage slice 0
  slice_copy<NT * 64>(wbuf, pw, tid);

  v8f acc0[NTH], acc1[NTH];
#pragma unroll
  for (int t = 0; t < NTH; ++t) {
    acc0[t] = (v8f){0.f,0.f,0.f,0.f,0.f,0.f,0.f,0.f};
    acc1[t] = (v8f){0.f,0.f,0.f,0.f,0.f,0.f,0.f,0.f};
  }

  const int mg = wave & 3;   // m-group (32 rows)
  const int nh = wave >> 2;  // n-half
  const int kb = (lane >> 4) * 8;
  // fold lane-dependent pieces into base pointers once; all loop offsets are
  // compile-time constants -> DS immediate offsets, no per-fragment VALU.
  const _Float16* arow0 = src + (mg * 32 + (lane & 15)) * SSTR + kb;
  const _Float16* arow1 = arow0 + 16 * SSTR;
  const unsigned int* bbase = wbuf + nh * (NTH * FRAG_DW) + lane * 8;

  slice_sync();

#pragma unroll
  for (int ks = 0; ks < KS; ++ks) {
    if (ks + 1 < KS)
      slice_copy<NT * 64>(wbuf + ((ks + 1) & 1) * (NT * FRAG_DW),
                          pw + (size_t)(ks + 1) * NT * FRAG_DW, tid);

    FragU a0, a1;
#pragma unroll
    for (int j = 0; j < 8; ++j) {
      const int koff = (j < 4) ? (2 * j) : (16 + 2 * (j - 4));  // constant
      a0.u[j] = *(const unsigned int*)(arow0 + ks * 32 + koff);
      a1.u[j] = *(const unsigned int*)(arow1 + ks * 32 + koff);
    }
#pragma unroll
    for (int t = 0; t < NTH; ++t) {
      const int fragoff = ((ks & 1) * NT + t) * FRAG_DW;        // constant
      FragU b;
      b.q[0] = *(const v4u*)(bbase + fragoff);
      b.q[1] = *(const v4u*)(bbase + fragoff + 4);
      acc0[t] = __builtin_amdgcn_wmma_f32_16x16x32_f16(
          false, a0.h, false, b.h, (short)0, acc0[t], false, false);
      acc1[t] = __builtin_amdgcn_wmma_f32_16x16x32_f16(
          false, a1.h, false, b.h, (short)0, acc1[t], false, false);
    }
    slice_sync();
  }

  // C layout: lanes 0-15 -> N=lane, M=r; lanes 16-31 -> N=lane-16, M=r+8
  const int mrow = mg * 32 + ((lane >> 4) << 3);
  const int nl   = lane & 15;
#pragma unroll
  for (int t = 0; t < NTH; ++t) {
    const int n = (nh * NTH + t) * 16 + nl;
    const float bv = bias[n];
#pragma unroll
    for (int r = 0; r < 8; ++r) {
      dst[(mrow + r) * DSTR + n]        = (_Float16)celu_f(acc0[t][r] + bv);
      dst[(mrow + 16 + r) * DSTR + n]   = (_Float16)celu_f(acc1[t][r] + bv);
    }
  }
}

// ---------------------------------------------------------------------------
// Main kernel: one workgroup = 128 atoms of one species through the full MLP.
// ---------------------------------------------------------------------------
__global__ void __launch_bounds__(256)
ani_mlp_kernel(const float* __restrict__ aev,
               const unsigned int* __restrict__ P1,
               const unsigned int* __restrict__ P2,
               const unsigned int* __restrict__ P3,
               const float* __restrict__ b1,
               const float* __restrict__ b2,
               const float* __restrict__ b3,
               const float* __restrict__ W4,
               const float* __restrict__ b4,
               const int* __restrict__ Hidx, const int* __restrict__ Cidx,
               const int* __restrict__ Nidx, const int* __restrict__ Oidx,
               int chunk, int tiles_per_spec,
               float* __restrict__ out) {
  __shared__ __align__(16) unsigned char lds0[MTILE * STRIDE_A * 2];   // A tile / h2
  __shared__ __align__(16) unsigned char lds1[MTILE * STRIDE_H1 * 2];  // h1 / h3
  __shared__ __align__(16) unsigned int  wbuf[WBUF_DW];                // weight slices / reduce

  const int tid  = threadIdx.x;
  const int lane = tid & 31;
  const int wave = tid >> 5;
  const int wg   = blockIdx.x;
  const int s    = wg / tiles_per_spec;
  const int tile = wg % tiles_per_spec;
  const int* idx = (s == 0) ? Hidx : (s == 1) ? Cidx : (s == 2) ? Nidx : Oidx;
  const int base = tile * MTILE;

  _Float16* sA = (_Float16*)lds0;
  _Float16* sH = (_Float16*)lds1;

  // ---- gather 128 AEV rows -> f16 tile in LDS (1 idx load + 48 b128/thread) ----
  {
    const int r    = tid >> 1;          // 2 threads per row
    const int half = tid & 1;
    const int g    = base + r;
    const bool valid = (g < chunk);
    const float* srcrow = valid ? (aev + (size_t)idx[g] * AEV_DIM) : aev;
    _Float16* drow = sA + r * STRIDE_A;
#pragma unroll 4
    for (int c4 = half * (AEV_DIM / 8); c4 < (half + 1) * (AEV_DIM / 8); ++c4) {
      float4 v = {0.f, 0.f, 0.f, 0.f};
      if (valid) v = *(const float4*)(srcrow + c4 * 4);
      union { _Float16 f[2]; unsigned int u; } p0, p1;
      p0.f[0] = (_Float16)v.x; p0.f[1] = (_Float16)v.y;
      p1.f[0] = (_Float16)v.z; p1.f[1] = (_Float16)v.w;
      *(unsigned int*)(drow + c4 * 4)     = p0.u;
      *(unsigned int*)(drow + c4 * 4 + 2) = p1.u;
    }
  }
  __syncthreads();

  // ---- layer 1: [128,384] x [384,160] ----
  mlp_layer<NT1, KS1, STRIDE_A, STRIDE_H1>(sA, sH, P1 + (size_t)s * KS1 * NT1 * FRAG_DW,
                                           wbuf, b1 + s * H1D, wave, lane, tid);
  __syncthreads();
  // ---- layer 2: [128,160] x [160,128] ----
  mlp_layer<NT2, KS2, STRIDE_H1, STRIDE_H2>(sH, sA, P2 + (size_t)s * KS2 * NT2 * FRAG_DW,
                                            wbuf, b2 + s * H2D, wave, lane, tid);
  __syncthreads();
  // ---- layer 3: [128,128] x [128,96] ----
  mlp_layer<NT3, KS3, STRIDE_H2, STRIDE_H3>(sA, sH, P3 + (size_t)s * KS3 * NT3 * FRAG_DW,
                                            wbuf, b3 + s * H3D, wave, lane, tid);
  __syncthreads();

  // ---- layer 4 (96 -> 1) + block reduction -> one atomic add ----
  float* red = (float*)wbuf;   // wbuf no longer needed
  float e = 0.f;
  if (tid < MTILE) {
    const int g = base + tid;
    if (g < chunk) {
      const _Float16* hp = sH + tid * STRIDE_H3;
      const float* w4 = W4 + s * H3D;
      float a = b4[s];
#pragma unroll
      for (int k = 0; k < H3D; ++k) a += (float)hp[k] * w4[k];
      e = a;
    }
    red[tid] = e;
  }
  __syncthreads();
  for (int off = 64; off > 0; off >>= 1) {
    if (tid < off) red[tid] += red[tid + off];
    __syncthreads();
  }
  if (tid == 0) atomicAdd(out, red[0]);
}

// ---------------------------------------------------------------------------
// Host launcher (graph-capture safe: only async ops on `stream`).
// ---------------------------------------------------------------------------
extern "C" void kernel_launch(void* const* d_in, const int* in_sizes, int n_in,
                              void* d_out, int out_size, void* d_ws, size_t ws_size,
                              hipStream_t stream) {
  const float* aev = (const float*)d_in[0];
  const float* W1  = (const float*)d_in[1];
  const float* b1  = (const float*)d_in[2];
  const float* W2  = (const float*)d_in[3];
  const float* b2  = (const float*)d_in[4];
  const float* W3  = (const float*)d_in[5];
  const float* b3  = (const float*)d_in[6];
  const float* W4  = (const float*)d_in[7];
  const float* b4  = (const float*)d_in[8];
  const int* Hi = (const int*)d_in[9];
  const int* Ci = (const int*)d_in[10];
  const int* Ni = (const int*)d_in[11];
  const int* Oi = (const int*)d_in[12];

  const int chunk = in_sizes[9];                    // atoms per species (50000)
  const int tiles = (chunk + MTILE - 1) / MTILE;    // 391

  unsigned int* P1 = (unsigned int*)d_ws;           // ~736 KB packed f16 weights
  unsigned int* P2 = P1 + DW_W1;
  unsigned int* P3 = P2 + DW_W2;

  (void)hipMemsetAsync(d_out, 0, sizeof(float), stream);

  const int totaldw = DW_W1 + DW_W2 + DW_W3;
  pack_weights_kernel<<<(totaldw + 255) / 256, 256, 0, stream>>>(W1, W2, W3, P1, P2, P3);

  ani_mlp_kernel<<<NSPEC * tiles, 256, 0, stream>>>(
      aev, P1, P2, P3, b1, b2, b3, W4, b4, Hi, Ci, Ni, Oi, chunk, tiles,
      (float*)d_out);
}
